// ShiftWiseConv_57458072486190
// MI455X (gfx1250) — matrix operand order repaired and checked by reading, not compiled, not run on hardware
//
#include <hip/hip_runtime.h>
#include <hip/hip_bf16.h>
#include <math.h>

typedef __attribute__((ext_vector_type(16))) __bf16 v16bf;
typedef __attribute__((ext_vector_type(8)))  __bf16 v8bf;
typedef __attribute__((ext_vector_type(4), aligned(2))) __bf16 v4bf_u; // unaligned view
typedef __attribute__((ext_vector_type(8)))  float  v8f;
typedef __attribute__((ext_vector_type(4)))  float  v4f;

#define C1   256
#define C2   256
#define KS   13
#define H    56
#define W    56
#define NB   16
#define PAD  6
#define HW   (H * W)          // 3136
#define KTOT (C1 * KS * KS)   // 43264, divisible by 32
#define NPIX (NB * H * W)     // 50176, divisible by 128

#define BM 256
#define BN 128
#define BK 32
#define NT 512                // 16 waves
#define LSTR 40               // padded LDS row stride in bf16 (80B)

#define W_ELEMS  ((size_t)C2 * KTOT)       // 11,075,584
#define X_ELEMS  ((size_t)NB * C1 * HW)    // 12,845,056
#define W_BYTES  (W_ELEMS * 2)             // 22,151,168
#define GUARD_B  1024                      // guard zone bytes around xbf (>684 needed)

// ---------------- one-shot f32 -> bf16 conversion (8 elems/thread) ----------
__global__ __launch_bounds__(256)
void cvt_f32_to_bf16(const float* __restrict__ src, __bf16* __restrict__ dst,
                     unsigned n8)
{
    unsigned i = blockIdx.x * 256u + threadIdx.x;
    if (i >= n8) return;
    size_t base = (size_t)i * 8;
    v4f f0 = *(const v4f*)(src + base);
    v4f f1 = *(const v4f*)(src + base + 4);
    v8bf o;
#pragma unroll
    for (int e = 0; e < 4; ++e) {
        o[e]     = (__bf16)f0[e];
        o[4 + e] = (__bf16)f1[e];
    }
    *(v8bf*)(dst + base) = o;
}

// ---------------- implicit-GEMM conv + BN + SiLU ----------------------------
__global__ __launch_bounds__(NT)
void conv_bn_silu_wmma(const __bf16* __restrict__ xbf,   // guard-padded
                       const __bf16* __restrict__ wbf,
                       const float* __restrict__ gamma,
                       const float* __restrict__ beta,
                       const float* __restrict__ rmean,
                       const float* __restrict__ rvar,
                       float* __restrict__ out)
{
    __shared__ __bf16 Abuf[2][BM * LSTR];   // weights tiles  [m][k], double-buffered
    __shared__ __bf16 Bbuf[2][BN * LSTR];   // im2col tiles   [n][k]

    const int tid    = threadIdx.x;
    const int lane   = tid & 31;
    const int wid    = tid >> 5;         // 0..15
    const int wave_m = wid & 7;          // 8 waves along M (32 rows each)
    const int wave_n = wid >> 3;         // 2 waves along N (64 cols each)
    const int half   = lane >> 4;
    const int l16    = lane & 15;

    const int m0 = blockIdx.x * BM;      // == 0 (grid.x = 1), kept general
    const int n0 = blockIdx.y * BN;

    v8f acc[2][4];
#pragma unroll
    for (int i = 0; i < 2; ++i)
#pragma unroll
        for (int j = 0; j < 4; ++j)
            acc[i][j] = (v8f)(0.0f);

    // ---- A loader: 2 threads per row, 16 contiguous bf16 each (256 rows) ----
    const int a_row = tid >> 1;
    const int a_col = (tid & 1) * 16;
    const __bf16* a_src0 = wbf + (size_t)(m0 + a_row) * KTOT + a_col;
    const int a_off = a_row * LSTR + a_col;

    // ---- B loader: 2 k-columns x 4 contiguous pixels per thread ----
    const int kpair = tid & 15;          // k = 2*kpair, 2*kpair+1
    const int seg   = (tid >> 4) * 4;    // pixel segment base (4 | HW)
    const int nseg  = n0 + seg;
    const int nb    = nseg / HW;
    const int rem0  = nseg - nb * HW;
    int ohw[4];                          // packed (oh<<16)|ow
#pragma unroll
    for (int i = 0; i < 4; ++i) {
        const int rem = rem0 + i;
        const int oh  = rem / W;
        ohw[i] = (oh << 16) | (rem - oh * W);
    }
    const int b_off = seg * LSTR + 2 * kpair;

    // incremental kk -> (c, kh, kw) for the even k = k0 + 2*kpair
    int c  = 0;
    int kh = (2 * kpair) / KS;
    int kw = (2 * kpair) - kh * KS;

    // ---- register staging (mask application deferred to store phase) ----
    v8bf  areg0, areg1;
    v4bf_u raw0, raw1;
    int khm6, kwm6, kh1m6, kw1m6;

    auto loadA = [&](int k0) {
        areg0 = *(const v8bf*)(a_src0 + k0);
        areg1 = *(const v8bf*)(a_src0 + k0 + 8);
    };
    auto loadB_and_advance = [&]() {
        khm6 = kh - PAD; kwm6 = kw - PAD;
        int kw1 = kw + 1, kh1 = kh, c1 = c;
        if (kw1 >= KS) { kw1 = 0; kh1 += 1; if (kh1 >= KS) { kh1 = 0; c1 += 1; } }
        kh1m6 = kh1 - PAD; kw1m6 = kw1 - PAD;
        // im2col address is linear in pixel index (valid across row wraps; guarded)
        const long base0 = (long)(nb * C1 + c)  * HW + rem0 + khm6  * W + kwm6;
        const long base1 = (long)(nb * C1 + c1) * HW + rem0 + kh1m6 * W + kw1m6;
        raw0 = *(const v4bf_u*)(xbf + base0);
        raw1 = *(const v4bf_u*)(xbf + base1);
        // advance even-k by BK=32: 32 = 2*13 + 6
        kw += 6; kh += 2;
        if (kw >= KS) { kw -= KS; kh += 1; }
        if (kh >= KS) { kh -= KS; c += 1; }
    };
    auto storeLDS = [&](int q) {
        __bf16* ad = &Abuf[q][a_off];
        *(v8bf*)(ad)     = areg0;
        *(v8bf*)(ad + 8) = areg1;
        __bf16* bd = &Bbuf[q][b_off];
#pragma unroll
        for (int i = 0; i < 4; ++i) {
            const int oh = ohw[i] >> 16, ow = ohw[i] & 0xffff;
            union { __bf16 h[2]; unsigned u; } pk;
            pk.h[0] = ((unsigned)(oh + khm6)  < (unsigned)H &&
                       (unsigned)(ow + kwm6)  < (unsigned)W) ? raw0[i] : (__bf16)0.0f;
            pk.h[1] = ((unsigned)(oh + kh1m6) < (unsigned)H &&
                       (unsigned)(ow + kw1m6) < (unsigned)W) ? raw1[i] : (__bf16)0.0f;
            *(unsigned*)(bd + i * LSTR) = pk.u;
        }
    };

    // prologue: stage tile k0 = 0 into buffer 0
    loadA(0);
    loadB_and_advance();
    storeLDS(0);
    __syncthreads();

    int p = 0;
    for (int k0 = BK; k0 <= KTOT; k0 += BK) {
        const bool more = (k0 < KTOT);
        if (more) {                      // prefetch next tile into registers
            loadA(k0);
            loadB_and_advance();
            __builtin_prefetch(a_src0 + k0 + BK, 0, 1);  // global_prefetch_b8
        }

        // ---- compute from LDS buffer p ----
        const __bf16* As = &Abuf[p][0];
        const __bf16* Bs = &Bbuf[p][0];
        v16bf afrag[2], bfrag[4];
#pragma unroll
        for (int i = 0; i < 2; ++i) {
            const int row = wave_m * 32 + i * 16 + l16;
            const __bf16* pp = As + row * LSTR + half * 8;
            v8bf lo = *(const v8bf*)(pp);
            v8bf hi = *(const v8bf*)(pp + 16);
#pragma unroll
            for (int e = 0; e < 8; ++e) { afrag[i][e] = lo[e]; afrag[i][8 + e] = hi[e]; }
        }
#pragma unroll
        for (int j = 0; j < 4; ++j) {
            const int col = wave_n * 64 + j * 16 + l16;
            const __bf16* pp = Bs + col * LSTR + half * 8;
            v8bf lo = *(const v8bf*)(pp);
            v8bf hi = *(const v8bf*)(pp + 16);
#pragma unroll
            for (int e = 0; e < 8; ++e) { bfrag[j][e] = lo[e]; bfrag[j][8 + e] = hi[e]; }
        }
#pragma unroll
        for (int i = 0; i < 2; ++i)
#pragma unroll
            for (int j = 0; j < 4; ++j)
                acc[i][j] = __builtin_amdgcn_wmma_f32_16x16x32_bf16(
                    false, afrag[i], false, bfrag[j],
                    (short)0, acc[i][j], false, false);

        // ---- write next tile into the other buffer, then one barrier ----
        if (more) storeLDS(p ^ 1);
        __syncthreads();
        p ^= 1;
    }

    // ---- epilogue: fused BatchNorm + SiLU, NCHW store ----
    // C/D layout: lane l, VGPR r -> M = r + 8*(l>=16), N = l%16
    float sv[2][8], bv[2][8];
#pragma unroll
    for (int i = 0; i < 2; ++i) {
        const int mbase = m0 + wave_m * 32 + i * 16 + 8 * half;
#pragma unroll
        for (int r = 0; r < 8; ++r) {
            const int m = mbase + r;
            const float s = gamma[m] * rsqrtf(rvar[m] + 1e-5f);
            sv[i][r] = s;
            bv[i][r] = beta[m] - rmean[m] * s;
        }
    }
#pragma unroll
    for (int i = 0; i < 2; ++i) {
        const int mbase = m0 + wave_m * 32 + i * 16 + 8 * half;
#pragma unroll
        for (int j = 0; j < 4; ++j) {
            const int n   = n0 + wave_n * 64 + j * 16 + l16;
            const int b   = n / HW;
            const int rem = n - b * HW;
            float* po = out + ((size_t)b * C2 + mbase) * HW + rem;
#pragma unroll
            for (int r = 0; r < 8; ++r) {
                const float y = acc[i][j][r] * sv[i][r] + bv[i][r];
                po[(size_t)r * HW] = y * (1.0f / (1.0f + __expf(-y)));
            }
        }
    }
}

extern "C" void kernel_launch(void* const* d_in, const int* in_sizes, int n_in,
                              void* d_out, int out_size, void* d_ws, size_t ws_size,
                              hipStream_t stream) {
    (void)in_sizes; (void)n_in; (void)out_size; (void)ws_size;
    const float* x     = (const float*)d_in[0];
    const float* w     = (const float*)d_in[1];
    const float* gamma = (const float*)d_in[2];
    const float* beta  = (const float*)d_in[3];
    const float* rmean = (const float*)d_in[4];
    const float* rvar  = (const float*)d_in[5];
    float* out = (float*)d_out;

    // scratch: [ w_bf16 22.15MB ][ 1KB guard ][ x_bf16 25.69MB ][ 1KB guard ]
    __bf16* wbf = (__bf16*)d_ws;
    __bf16* xbf = (__bf16*)((char*)d_ws + W_BYTES + GUARD_B);

    unsigned w8 = (unsigned)(W_ELEMS / 8);
    unsigned x8 = (unsigned)(X_ELEMS / 8);
    cvt_f32_to_bf16<<<dim3((w8 + 255) / 256), 256, 0, stream>>>(w, wbf, w8);
    cvt_f32_to_bf16<<<dim3((x8 + 255) / 256), 256, 0, stream>>>(x, xbf, x8);

    dim3 grid(C2 / BM, NPIX / BN);   // 1 x 392
    conv_bn_silu_wmma<<<grid, NT, 0, stream>>>(xbf, wbf, gamma, beta,
                                               rmean, rvar, out);
}
